// VectorQuantizedVariationalAutoEncoder_49615462203805
// MI455X (gfx1250) — compile-verified, matrix operand order checked by reading, not run on hardware
//
#include <hip/hip_runtime.h>
#include <hip/hip_bf16.h>
#include <math.h>

typedef __attribute__((ext_vector_type(4)))  float  v4f;
typedef __attribute__((ext_vector_type(8)))  float  v8f;
typedef __attribute__((ext_vector_type(4)))  __bf16 v4bf;
typedef __attribute__((ext_vector_type(8)))  __bf16 v8bf;
typedef __attribute__((ext_vector_type(16))) __bf16 v16bf;

__device__ __forceinline__ __bf16 to_bf16(float f) { return (__bf16)f; }
__device__ __forceinline__ v16bf concat8(v8bf lo, v8bf hi) {
    return __builtin_shufflevector(lo, hi, 0, 1, 2, 3, 4, 5, 6, 7,
                                   8, 9, 10, 11, 12, 13, 14, 15);
}

// ---------------------------------------------------------------------------
// Implicit-GEMM WMMA kernel.  C(M,N) = A_gather(M,K) x B(K,N) [+ bias]
//   MODE 0: plain dense A (lda = Kreal)
//   MODE 1: conv (implicit im2col), dense NHWC store
//   MODE 2: convT parity class (2x2 taps via LUT), parity-scattered store
//   ORDER 0: k = tap*Cin + c (Cin pow2; aligned 16-chunk = one tap, v4f loads)
//   ORDER 1: k = c*16 + tap  (ec1: Cin=3, KH*KW=16; scalar gather)
// Tap geometry is a 4-bit packed LUT (khPack/kwPack) -> no division anywhere
// in the hot loop; all addressing is shift/mask (every dim is a pow2).
// B is pre-packed into WMMA-fragment tiles: tile(kt,nt) base =
// (kt*NtilesN+nt)*512, element = lane*16+j -> one 32B contiguous load per frag.
// Block tile 128x64, 8 waves, 2x2 WMMA tiles per wave; A tile double-buffered
// in LDS (one barrier per K-step).
// ---------------------------------------------------------------------------
struct GemmGeom {
    const float*  A;
    const __bf16* Bm;
    const float*  bias;
    float*        C;
    int M, N, Kreal, Kpad;
    int Hin, Win, Cin;
    int stride, pad;
    int ph, pw;                    // convT parity (store)
    int HoutF, WoutF, ldc;         // store geometry
    int wShift, hwShift;           // log2(Wout), log2(Hout*Wout)
    int winShift, clog2;           // log2(Win), log2(Cin)
    int NtilesN;                   // Npad/16
    int reluIn;
    unsigned long long khPack, kwPack;  // 4-bit per-tap kh/kw LUT
};

#define LDA 40   // bf16 per LDS row (80B: 16B-aligned rows for ds_load_b128)

template <int MODE, int ORDER>
__device__ __forceinline__ void stageA(const GemmGeom& g, __bf16 (*sbuf)[LDA],
                                       int row, int kc, const float* rowBase,
                                       int ohs, int ows, int k0) {
    if (MODE == 0) {
        const v4f* src = (const v4f*)(rowBase + k0);
#pragma unroll
        for (int q = 0; q < 4; ++q) {
            v4f vv = src[q];
            v4bf hv = {to_bf16(vv[0]), to_bf16(vv[1]), to_bf16(vv[2]), to_bf16(vv[3])};
            *(v4bf*)&sbuf[row][kc + q * 4] = hv;
        }
    } else if (ORDER == 0) {
        int k   = k0 + kc;
        int c0  = k & ((1 << g.clog2) - 1);
        int tap = k >> g.clog2;
        int kh  = (int)((g.khPack >> (tap * 4)) & 15);
        int kw  = (int)((g.kwPack >> (tap * 4)) & 15);
        int ih  = ohs + kh, iw = ows + kw;
        bool ok = ((unsigned)ih < (unsigned)g.Hin) &
                  ((unsigned)iw < (unsigned)g.Win) & (k < g.Kreal);
        const v4f* src = (const v4f*)(rowBase +
                           ((((ih << g.winShift) + iw) << g.clog2) + c0));
        v4f z = {};
#pragma unroll
        for (int q = 0; q < 4; ++q) {
            v4f vv = ok ? src[q] : z;
            if (g.reluIn) {
                vv[0] = fmaxf(vv[0], 0.f); vv[1] = fmaxf(vv[1], 0.f);
                vv[2] = fmaxf(vv[2], 0.f); vv[3] = fmaxf(vv[3], 0.f);
            }
            v4bf hv = {to_bf16(vv[0]), to_bf16(vv[1]), to_bf16(vv[2]), to_bf16(vv[3])};
            *(v4bf*)&sbuf[row][kc + q * 4] = hv;
        }
    } else {  // ORDER 1: ec1 (Cin=3, 4x4 taps) — scalar, negligible FLOPs
#pragma unroll
        for (int j = 0; j < 16; ++j) {
            int k = k0 + kc + j;
            int c = k >> 4, tap = k & 15;
            float v = 0.f;
            if (c < g.Cin) {
                int kh = tap >> 2, kw = tap & 3;
                int ih = ohs + kh, iw = ows + kw;
                if ((unsigned)ih < (unsigned)g.Hin &&
                    (unsigned)iw < (unsigned)g.Win)
                    v = rowBase[((ih << g.winShift) + iw) * g.Cin + c];
            }
            sbuf[row][kc + j] = to_bf16(v);
        }
    }
}

template <int MODE, int ORDER>
__global__ __launch_bounds__(256) void gemm_wmma(GemmGeom g) {
    __shared__ __bf16 sA[2][128][LDA];

    const int t    = threadIdx.x;
    const int wave = t >> 5;
    const int lane = t & 31;
    const int bm   = blockIdx.x * 128;
    const int bn   = blockIdx.y * 64;
    const int mg   = wave >> 1;   // m-tiles {2mg, 2mg+1}
    const int ngrp = wave & 1;    // n-tiles {2ngrp, 2ngrp+1}

    const int lane15 = lane & 15;
    const int kb     = (lane & 16) ? 8 : 0;  // ISA 7.12.2 K-split by lane half

    // ---- per-thread A-staging setup: one row, one 16-wide K chunk ----
    const int row = t >> 1;
    const int kc  = (t & 1) << 4;
    const int m   = bm + row;                 // M is a multiple of 128
    int ohs = 0, ows = 0;
    const float* rowBase;
    if (MODE == 0) {
        rowBase = g.A + (size_t)m * g.Kreal + kc;
    } else {
        int bb  = m >> g.hwShift;
        int rem = m & ((1 << g.hwShift) - 1);
        int oh  = rem >> g.wShift;
        int ow  = rem & ((1 << g.wShift) - 1);
        ohs = oh * g.stride - g.pad;          // MODE 2 uses stride=1,pad=1
        ows = ow * g.stride - g.pad;
        rowBase = g.A + (size_t)bb * ((size_t)g.Hin << g.winShift << g.clog2);
    }

    v8f acc00 = {}, acc01 = {}, acc10 = {}, acc11 = {};

    const int arow0 = (mg * 2) * 16 + lane15;
    const int arow1 = arow0 + 16;
    const int ct0   = (bn >> 4) + ngrp * 2;   // global n-tile of b0

    const int nsteps = g.Kpad >> 5;
    stageA<MODE, ORDER>(g, sA[0], row, kc, rowBase, ohs, ows, 0);
    __syncthreads();

    for (int s = 0; s < nsteps; ++s) {
        const int cur = s & 1;
        if (s + 1 < nsteps)
            stageA<MODE, ORDER>(g, sA[cur ^ 1], row, kc, rowBase, ohs, ows,
                                (s + 1) << 5);

        v16bf a0 = concat8(*(const v8bf*)&sA[cur][arow0][kb],
                           *(const v8bf*)&sA[cur][arow0][kb + 16]);
        v16bf a1 = concat8(*(const v8bf*)&sA[cur][arow1][kb],
                           *(const v8bf*)&sA[cur][arow1][kb + 16]);
        size_t bbase = ((size_t)s * g.NtilesN + ct0) << 9;
        v16bf b0 = *(const v16bf*)(g.Bm + bbase + lane * 16);
        v16bf b1 = *(const v16bf*)(g.Bm + bbase + 512 + lane * 16);

        acc00 = __builtin_amdgcn_wmma_f32_16x16x32_bf16(false, a0, false, b0,
                                                        (short)0, acc00, false, false);
        acc01 = __builtin_amdgcn_wmma_f32_16x16x32_bf16(false, a0, false, b1,
                                                        (short)0, acc01, false, false);
        acc10 = __builtin_amdgcn_wmma_f32_16x16x32_bf16(false, a1, false, b0,
                                                        (short)0, acc10, false, false);
        acc11 = __builtin_amdgcn_wmma_f32_16x16x32_bf16(false, a1, false, b1,
                                                        (short)0, acc11, false, false);
        __syncthreads();   // publishes next tile, protects the one just read
    }

    // ================= epilogue =================
    const int rowoff = (lane & 16) ? 8 : 0;
#pragma unroll
    for (int mi = 0; mi < 2; ++mi) {
#pragma unroll
        for (int ni = 0; ni < 2; ++ni) {
            v8f acc = (mi == 0) ? (ni == 0 ? acc00 : acc01)
                                : (ni == 0 ? acc10 : acc11);
            int nbase = bn + (ngrp * 2 + ni) * 16 + lane15;
            if (nbase >= g.N) continue;
            float bv = g.bias ? g.bias[nbase] : 0.f;
            int mbase = bm + (mg * 2 + mi) * 16 + rowoff;
#pragma unroll
            for (int r = 0; r < 8; ++r) {
                int mgr = mbase + r;
                float v = acc[r] + bv;
                size_t off;
                if (MODE == 2) {
                    int bb  = mgr >> g.hwShift;
                    int rem = mgr & ((1 << g.hwShift) - 1);
                    int oh  = rem >> g.wShift;
                    int ow  = rem & ((1 << g.wShift) - 1);
                    off = (((size_t)bb * g.HoutF + (2 * oh + g.ph)) * g.WoutF +
                           (2 * ow + g.pw)) * g.ldc + nbase;
                } else {
                    off = (size_t)mgr * g.ldc + nbase;
                }
                g.C[off] = v;
            }
        }
    }
}

// ---------------------------------------------------------------------------
// Weight packing into WMMA-fragment tile layout (fp32 -> bf16)
// ---------------------------------------------------------------------------
__device__ __forceinline__ void frag_kn(int i, int NtilesN, int& k, int& n) {
    int tile = i >> 9, w = i & 511, lane = w >> 4, j = w & 15;
    int kt = tile / NtilesN, nt = tile - kt * NtilesN;
    k = kt * 32 + ((lane & 16) ? 8 : 0) + (j < 8 ? j : j + 8);
    n = nt * 16 + (lane & 15);
}

__global__ void pack_conv_frag(const float* __restrict__ w, __bf16* __restrict__ Bp,
                               int Cout, int Cin, int KH, int KW, int Kreal,
                               int total, int NtilesN, int order, int clog2) {
    int i = blockIdx.x * blockDim.x + threadIdx.x;
    if (i >= total) return;
    int k, n; frag_kn(i, NtilesN, k, n);
    float v = 0.f;
    if (k < Kreal && n < Cout) {
        int c, kh, kw;
        if (order == 0) { c = k & ((1 << clog2) - 1); int r = k >> clog2;
                          kh = r / KW; kw = r - kh * KW; }
        else            { int tap = k & 15; c = k >> 4; kh = tap >> 2; kw = tap & 3; }
        if (c < Cin) v = w[(((size_t)n * Cin + c) * KH + kh) * KW + kw];  // OIHW
    }
    Bp[i] = to_bf16(v);
}

__global__ void pack_convt_frag(const float* __restrict__ w, __bf16* __restrict__ Bp,
                                int Cout, int Cin, int ph, int pw,
                                int total, int NtilesN) {
    int i = blockIdx.x * blockDim.x + threadIdx.x;
    if (i >= total) return;
    int k, n; frag_kn(i, NtilesN, k, n);
    float v = 0.f;
    if (n < Cout) {
        int c = k & (Cin - 1), tap = k / Cin;
        int th = tap >> 1, tw = tap & 1;
        int kh = 2 * th + (1 - ph), kw = 2 * tw + (1 - pw);
        v = w[(((size_t)c * Cout + n) * 4 + kh) * 4 + kw];   // (Cin,Cout,4,4)
    }
    Bp[i] = to_bf16(v);
}

__global__ void pack_emb_frag(const float* __restrict__ E, __bf16* __restrict__ Bp,
                              int D, int Kc, int total, int NtilesN) {
    int i = blockIdx.x * blockDim.x + threadIdx.x;
    if (i >= total) return;
    int k, n; frag_kn(i, NtilesN, k, n);
    Bp[i] = to_bf16(E[(size_t)n * D + k]);   // B[k][n] = E[n][k]
}

// ---------------------------------------------------------------------------
// BatchNorm (training-mode batch stats) + elementwise helpers
// ---------------------------------------------------------------------------
__global__ void zero_f32(float* p, int n) {
    int i = blockIdx.x * blockDim.x + threadIdx.x;
    if (i < n) p[i] = 0.0f;
}

__global__ void bn_stats(const float* __restrict__ x, float* __restrict__ stats,
                         int M, int C) {   // blockDim.x == C == 256
    int c  = threadIdx.x;
    int r0 = blockIdx.x * 256;
    int r1 = r0 + 256; if (r1 > M) r1 = M;
    float s = 0.0f, ss = 0.0f;
    for (int r = r0; r < r1; ++r) {
        float v = x[(size_t)r * C + c];
        s += v; ss += v * v;
    }
    atomicAdd(&stats[c], s);
    atomicAdd(&stats[C + c], ss);
}

__global__ void bn_apply(const float* __restrict__ x, const float* __restrict__ res,
                         float* __restrict__ y, const float* __restrict__ stats,
                         const float* __restrict__ gm, const float* __restrict__ bt,
                         int total, float invM, int relu) {
    int i = blockIdx.x * blockDim.x + threadIdx.x;
    if (i >= total) return;
    int c = i & 255;                       // C == 256
    float mean = stats[c] * invM;
    float var  = stats[256 + c] * invM - mean * mean;
    float v = (x[i] - mean) * rsqrtf(var + 1e-5f) * gm[c] + bt[c];
    if (relu) v = fmaxf(v, 0.0f);
    if (res) v += res[i];
    y[i] = v;
}

__global__ void nchw_to_nhwc(const float* __restrict__ in, float* __restrict__ out,
                             int B, int C, int H, int W) {
    int i = blockIdx.x * blockDim.x + threadIdx.x;
    if (i >= B * C * H * W) return;                  // i indexes NHWC output
    int c = i % C; int r = i / C;
    int w = r % W; r /= W;
    int h = r % H; int b = r / H;
    out[i] = in[(((size_t)b * C + c) * H + h) * W + w];
}

__global__ void nhwc_to_nchw(const float* __restrict__ in, float* __restrict__ out,
                             int B, int C, int H, int W, int sigm) {
    int i = blockIdx.x * blockDim.x + threadIdx.x;
    if (i >= B * C * H * W) return;                  // i indexes NCHW output
    int w = i % W; int r = i / W;
    int h = r % H; r /= H;
    int c = r % C; int b = r / C;
    float v = in[(((size_t)b * H + h) * W + w) * C + c];
    if (sigm) v = 1.0f / (1.0f + expf(-v));
    out[i] = v;
}

// ---------------------------------------------------------------------------
// Vector quantizer helpers
// ---------------------------------------------------------------------------
__global__ void sq_norm(const float* __restrict__ E, float* __restrict__ en,
                        int K, int D) {
    int n = blockIdx.x * blockDim.x + threadIdx.x;
    if (n >= K) return;
    float s = 0.0f;
    for (int d = 0; d < D; ++d) { float v = E[(size_t)n * D + d]; s += v * v; }
    en[n] = s;
}

__global__ void argmin_rows(const float* __restrict__ G, const float* __restrict__ en,
                            int* __restrict__ idx, int M, int N) {
    int wave = threadIdx.x >> 5;
    int lane = threadIdx.x & 31;
    int row  = blockIdx.x * 8 + wave;
    if (row >= M) return;
    float best = 3.4e38f; int bi = 0;
    for (int n = lane; n < N; n += 32) {
        float v = en[n] - 2.0f * G[(size_t)row * N + n];
        if (v < best) { best = v; bi = n; }
    }
    for (int off = 16; off > 0; off >>= 1) {
        float ov = __shfl_down(best, off, 32);
        int   oi = __shfl_down(bi,   off, 32);
        if (ov < best || (ov == best && oi < bi)) { best = ov; bi = oi; }
    }
    if (lane == 0) idx[row] = bi;
}

__global__ void gather_codes(const float* __restrict__ E, const int* __restrict__ idx,
                             float* __restrict__ zq, int M, int D) {
    int i = blockIdx.x * blockDim.x + threadIdx.x;
    if (i >= M * D) return;
    int m = i >> 8, d = i & 255;                     // D == 256
    zq[i] = E[(size_t)idx[m] * D + d];
}

// ---------------------------------------------------------------------------
// Host orchestration
// ---------------------------------------------------------------------------
static inline int ceil_div(int a, int b) { return (a + b - 1) / b; }

extern "C" void kernel_launch(void* const* d_in, const int* in_sizes, int n_in,
                              void* d_out, int out_size, void* d_ws, size_t ws_size,
                              hipStream_t stream) {
    (void)in_sizes; (void)n_in; (void)out_size; (void)ws_size;

    const float* x_in  = (const float*)d_in[0];
    const float* ec1w  = (const float*)d_in[1];
    const float* ec1b  = (const float*)d_in[2];
    const float* ebn_g = (const float*)d_in[3];
    const float* ebn_b = (const float*)d_in[4];
    const float* ec2w  = (const float*)d_in[5];
    const float* ec2b  = (const float*)d_in[6];
    const int rb_base[4] = {7, 15, 24, 32};          // erb1, erb2, drb1, drb2
    const float* emb   = (const float*)d_in[23];
    const float* dt1w  = (const float*)d_in[40];
    const float* dt1b  = (const float*)d_in[41];
    const float* dbn_g = (const float*)d_in[42];
    const float* dbn_b = (const float*)d_in[43];
    const float* dt2w  = (const float*)d_in[44];
    const float* dt2b  = (const float*)d_in[45];

    const int B = 32, D = 256, Kc = 512;

    size_t off = 0;
    auto alloc = [&](size_t bytes) -> void* {
        off = (off + 255) & ~(size_t)255;
        void* p = (char*)d_ws + off;
        off += bytes;
        return p;
    };
    float* a_x  = (float*)alloc((size_t)B * 128 * 128 * 3 * 4);
    float* big  = (float*)alloc((size_t)B * 64 * 64 * D * 4);
    float* h2   = (float*)alloc((size_t)B * 32 * 32 * D * 4);
    float* t1   = (float*)alloc((size_t)B * 32 * 32 * D * 4);
    float* t2   = (float*)alloc((size_t)B * 32 * 32 * D * 4);
    float* t3   = (float*)alloc((size_t)B * 32 * 32 * D * 4);
    float* ze   = (float*)alloc((size_t)B * 32 * 32 * D * 4);
    float* zq   = (float*)alloc((size_t)B * 32 * 32 * D * 4);
    float* G    = (float*)alloc((size_t)32768 * Kc * 4);
    int*   idx  = (int*)  alloc((size_t)32768 * 4);
    float* d4   = (float*)alloc((size_t)B * 128 * 128 * 3 * 4);
    float* stats= (float*)alloc(2 * 256 * 4);
    float* en   = (float*)alloc(Kc * 4);
    __bf16* ec1B = (__bf16*)alloc((size_t)64 * 256 * 2);
    __bf16* ec2B = (__bf16*)alloc((size_t)4096 * 256 * 2);
    __bf16* rbC3[4]; __bf16* rbC1[4];
    for (int i = 0; i < 4; ++i) {
        rbC3[i] = (__bf16*)alloc((size_t)2304 * 256 * 2);
        rbC1[i] = (__bf16*)alloc((size_t)256 * 256 * 2);
    }
    __bf16* embB = (__bf16*)alloc((size_t)256 * 512 * 2);
    __bf16* dt1B[4]; __bf16* dt2B[4];
    for (int p = 0; p < 4; ++p) dt1B[p] = (__bf16*)alloc((size_t)1024 * 256 * 2);
    for (int p = 0; p < 4; ++p) dt2B[p] = (__bf16*)alloc((size_t)1024 * 64 * 2);

    float* o_xhat = (float*)d_out;
    float* o_ze   = o_xhat + (size_t)B * 3 * 128 * 128;
    float* o_zq   = o_ze   + (size_t)B * D * 32 * 32;

    // ---- pack weights into fragment-tile layout ----
    auto packc = [&](const float* w, __bf16* Bp, int Cout, int Cin, int KH, int KW,
                     int Kreal, int Kpad, int Npad, int order, int clog2) {
        int tot = Kpad * Npad;
        pack_conv_frag<<<ceil_div(tot, 256), 256, 0, stream>>>(
            w, Bp, Cout, Cin, KH, KW, Kreal, tot, Npad >> 4, order, clog2);
    };
    packc(ec1w, ec1B, 256, 3, 4, 4, 48, 64, 256, /*order=*/1, 0);
    packc(ec2w, ec2B, 256, 256, 4, 4, 4096, 4096, 256, 0, 8);
    for (int i = 0; i < 4; ++i) {
        packc((const float*)d_in[rb_base[i] + 0], rbC3[i], 256, 256, 3, 3,
              2304, 2304, 256, 0, 8);
        packc((const float*)d_in[rb_base[i] + 4], rbC1[i], 256, 256, 1, 1,
              256, 256, 256, 0, 8);
    }
    {
        int tot = 256 * 512;
        pack_emb_frag<<<ceil_div(tot, 256), 256, 0, stream>>>(emb, embB, 256, Kc,
                                                              tot, Kc >> 4);
    }
    for (int p = 0; p < 4; ++p) {
        int tot1 = 1024 * 256;
        pack_convt_frag<<<ceil_div(tot1, 256), 256, 0, stream>>>(
            dt1w, dt1B[p], 256, 256, p >> 1, p & 1, tot1, 16);
        int tot2 = 1024 * 64;
        pack_convt_frag<<<ceil_div(tot2, 256), 256, 0, stream>>>(
            dt2w, dt2B[p], 3, 256, p >> 1, p & 1, tot2, 4);
    }

    // ---- tap LUT builders (4-bit packed) ----
    auto conv_lut = [](int KH, int KW, unsigned long long& khP, unsigned long long& kwP) {
        khP = 0; kwP = 0;
        for (int tap = 0; tap < KH * KW; ++tap) {
            khP |= (unsigned long long)(tap / KW) << (4 * tap);
            kwP |= (unsigned long long)(tap % KW) << (4 * tap);
        }
    };
    auto convt_lut = [](int ph, int pw, unsigned long long& khP, unsigned long long& kwP) {
        khP = 0; kwP = 0;
        for (int tap = 0; tap < 4; ++tap) {
            int th = tap >> 1, tw = tap & 1;
            int dh = (ph ? (1 - th) : -th) + 1;   // +1 bias; ohs = oh - 1
            int dw = (pw ? (1 - tw) : -tw) + 1;
            khP |= (unsigned long long)dh << (4 * tap);
            kwP |= (unsigned long long)dw << (4 * tap);
        }
    };

    auto lg = [](int v) { int s = 0; while ((1 << s) < v) ++s; return s; };
    auto conv = [&](const float* A, const __bf16* Bp, const float* bias, float* C,
                    int Hin, int Win, int Cin, int KH, int KW, int stride, int pad,
                    int Hout, int Wout, int N, int Npad, int Kreal, int Kpad,
                    int order, int reluIn) {
        GemmGeom g{};
        g.A = A; g.Bm = Bp; g.bias = bias; g.C = C;
        g.M = B * Hout * Wout; g.N = N; g.Kreal = Kreal; g.Kpad = Kpad;
        g.Hin = Hin; g.Win = Win; g.Cin = Cin;
        g.stride = stride; g.pad = pad;
        g.ldc = N; g.reluIn = reluIn;
        g.wShift = lg(Wout); g.hwShift = lg(Hout * Wout);
        g.winShift = lg(Win); g.clog2 = lg(Cin);
        g.NtilesN = Npad >> 4;
        conv_lut(KH, KW, g.khPack, g.kwPack);
        dim3 grid(g.M / 128, ceil_div(N, 64));
        if (order == 0) gemm_wmma<1, 0><<<grid, 256, 0, stream>>>(g);
        else            gemm_wmma<1, 1><<<grid, 256, 0, stream>>>(g);
    };
    auto convt = [&](const float* A, const __bf16* Bp, const float* bias, float* C,
                     int Hin, int N, int Npad, int ph, int pw, int reluIn) {
        GemmGeom g{};
        g.A = A; g.Bm = Bp; g.bias = bias; g.C = C;
        g.M = B * Hin * Hin; g.N = N; g.Kreal = 1024; g.Kpad = 1024;
        g.Hin = Hin; g.Win = Hin; g.Cin = 256;
        g.stride = 1; g.pad = 1;              // ohs = oh - 1 (LUT is +1 biased)
        g.ph = ph; g.pw = pw;
        g.HoutF = 2 * Hin; g.WoutF = 2 * Hin; g.ldc = N;
        g.wShift = lg(Hin); g.hwShift = lg(Hin * Hin);
        g.winShift = lg(Hin); g.clog2 = 8;
        g.NtilesN = Npad >> 4; g.reluIn = reluIn;
        convt_lut(ph, pw, g.khPack, g.kwPack);
        dim3 grid(g.M / 128, ceil_div(N, 64));
        gemm_wmma<2, 0><<<grid, 256, 0, stream>>>(g);
    };
    auto bnorm = [&](const float* xin, float* yout, const float* gm, const float* bt,
                     int M, int relu, const float* residual) {
        zero_f32<<<2, 256, 0, stream>>>(stats, 512);
        bn_stats<<<ceil_div(M, 256), 256, 0, stream>>>(xin, stats, M, 256);
        int tot = M * 256;
        bn_apply<<<ceil_div(tot, 256), 256, 0, stream>>>(
            xin, residual, yout, stats, gm, bt, tot, 1.0f / (float)M, relu);
    };
    auto resblock = [&](const float* xin, float* xout, int pb) {
        const float* c3b_ = (const float*)d_in[pb + 1];
        const float* bn1g = (const float*)d_in[pb + 2];
        const float* bn1b = (const float*)d_in[pb + 3];
        const float* c1b_ = (const float*)d_in[pb + 5];
        const float* bn2g = (const float*)d_in[pb + 6];
        const float* bn2b = (const float*)d_in[pb + 7];
        int rbi = (pb == 7) ? 0 : (pb == 15) ? 1 : (pb == 24) ? 2 : 3;
        conv(xin, rbC3[rbi], c3b_, t1, 32, 32, 256, 3, 3, 1, 1, 32, 32,
             256, 256, 2304, 2304, 0, /*reluIn=*/1);           // conv3(relu(x))
        bnorm(t1, t1, bn1g, bn1b, 32768, /*relu=*/1, nullptr); // relu(bn1)
        conv(t1, rbC1[rbi], c1b_, t3, 32, 32, 256, 1, 1, 1, 0, 32, 32,
             256, 256, 256, 256, 0, /*reluIn=*/0);             // conv1
        bnorm(t3, xout, bn2g, bn2b, 32768, /*relu=*/0, xin);   // bn2 + residual
    };

    // ================= forward pass =================
    {
        int tot = B * 3 * 128 * 128;
        nchw_to_nhwc<<<ceil_div(tot, 256), 256, 0, stream>>>(x_in, a_x, B, 3, 128, 128);
    }
    conv(a_x, ec1B, ec1b, big, 128, 128, 3, 4, 4, 2, 1, 64, 64,
         256, 256, 48, 64, /*order=*/1, 0);                    // (B,64,64,256)
    bnorm(big, big, ebn_g, ebn_b, B * 64 * 64, 1, nullptr);
    conv(big, ec2B, ec2b, h2, 64, 64, 256, 4, 4, 2, 1, 32, 32,
         256, 256, 4096, 4096, 0, 0);                          // (B,32,32,256)
    resblock(h2, t2, rb_base[0]);
    resblock(t2, ze, rb_base[1]);                              // z_e (NHWC)

    // ---- vector quantizer ----
    {
        GemmGeom g{};
        g.A = ze; g.Bm = embB; g.bias = nullptr; g.C = G;
        g.M = 32768; g.N = Kc; g.Kreal = 256; g.Kpad = 256;
        g.ldc = Kc; g.NtilesN = Kc >> 4;
        dim3 grid(g.M / 128, Kc / 64);
        gemm_wmma<0, 0><<<grid, 256, 0, stream>>>(g);          // G = f @ E^T
    }
    sq_norm<<<2, 256, 0, stream>>>(emb, en, Kc, 256);
    argmin_rows<<<32768 / 8, 256, 0, stream>>>(G, en, idx, 32768, Kc);
    gather_codes<<<ceil_div(32768 * 256, 256), 256, 0, stream>>>(emb, idx, zq, 32768, 256);
    {
        int tot = B * D * 32 * 32;
        nhwc_to_nchw<<<ceil_div(tot, 256), 256, 0, stream>>>(ze, o_ze, B, D, 32, 32, 0);
        nhwc_to_nchw<<<ceil_div(tot, 256), 256, 0, stream>>>(zq, o_zq, B, D, 32, 32, 0);
    }

    // ---- decoder (straight-through forward value == z_q) ----
    resblock(zq, h2, rb_base[2]);
    resblock(h2, t2, rb_base[3]);
    for (int p = 0; p < 4; ++p)                                 // convT1 (+fused relu)
        convt(t2, dt1B[p], dt1b, big, 32, 256, 256, p >> 1, p & 1, 1);
    bnorm(big, big, dbn_g, dbn_b, B * 64 * 64, 1, nullptr);
    for (int p = 0; p < 4; ++p)                                 // convT2 -> (B,128,128,3)
        convt(big, dt2B[p], dt2b, d4, 64, 3, 64, p >> 1, p & 1, 0);
    {
        int tot = B * 3 * 128 * 128;
        nhwc_to_nchw<<<ceil_div(tot, 256), 256, 0, stream>>>(d4, o_xhat, B, 3, 128, 128, 1);
    }
}
